// MoEAbstract_84250078478716
// MI455X (gfx1250) — compile-verified, matrix operand order checked by reading, not run on hardware
//
#include <hip/hip_runtime.h>
#include <hip/hip_bf16.h>

typedef __attribute__((ext_vector_type(16))) __bf16 v16bf;
typedef __attribute__((ext_vector_type(8)))  __bf16 v8bf;
typedef __attribute__((ext_vector_type(4)))  __bf16 v4bf;
typedef __attribute__((ext_vector_type(8)))  float  v8f;

#define B_ 1024
#define N_ 64
#define F_ 512
#define H_ 1024
#define E_ 8
#define D_ 2048
#define A_ 256

// ---------------------------------------------------------------------------
// Tensor Data Mover support (guarded; falls back to direct global loads)
// ---------------------------------------------------------------------------
#if defined(__has_builtin)
#  if __has_builtin(__builtin_amdgcn_tensor_load_to_lds) && \
      __has_builtin(__builtin_amdgcn_s_wait_tensorcnt)
#    define USE_TDM 1
#  endif
#endif
#ifndef USE_TDM
#  define USE_TDM 0
#endif

#if USE_TDM
typedef unsigned u32x4 __attribute__((ext_vector_type(4)));
typedef int      i32x4 __attribute__((ext_vector_type(4)));
typedef int      i32x8 __attribute__((ext_vector_type(8)));

// 2-D bf16 tile load via TDM: tile_d0 contiguous elements x tile_d1 rows,
// row stride = stride0 elements. D# packed per CDNA5 ISA 8.3/8.4.
__device__ __forceinline__ void tdm_load_2d_bf16(
    unsigned lds_off, const void* gptr,
    unsigned tensor_d0, unsigned tensor_d1, unsigned stride0,
    unsigned tile_d0, unsigned tile_d1) {
  const unsigned long long ga = (unsigned long long)gptr;
  u32x4 g0;
  g0[0] = 1u;                                                  // count=1
  g0[1] = lds_off;                                             // lds_addr
  g0[2] = (unsigned)ga;                                        // global_addr[31:0]
  g0[3] = (unsigned)((ga >> 32) & 0x01FFFFFFu) | (2u << 30);   // addr[56:32]|type=2
  i32x8 g1;
  g1[0] = (int)(1u << 16);                                     // data_size=1 (2B)
  g1[1] = (int)((tensor_d0 & 0xFFFFu) << 16);                  // tensor_dim0 lo
  g1[2] = (int)((tensor_d0 >> 16) & 0xFFFFu) |
          (int)((tensor_d1 & 0xFFFFu) << 16);                  // dim0 hi | dim1 lo
  g1[3] = (int)((tensor_d1 >> 16) & 0xFFFFu) |
          (int)((tile_d0 & 0xFFFFu) << 16);                    // dim1 hi | tile_dim0
  g1[4] = (int)(tile_d1 & 0xFFFFu);                            // tile_dim1 (dim2=0)
  g1[5] = (int)stride0;                                        // dim0_stride lo
  g1[6] = 0;
  g1[7] = 0;
  const i32x4 z4 = {0, 0, 0, 0};
#if __clang_major__ >= 23
  const i32x8 z8 = {0, 0, 0, 0, 0, 0, 0, 0};
  __builtin_amdgcn_tensor_load_to_lds(g0, g1, z4, z4, z8, 0);
#else
  __builtin_amdgcn_tensor_load_to_lds(g0, g1, z4, z4, 0);
#endif
}
#endif  // USE_TDM

// ---------------------------------------------------------------------------
// WMMA helpers (CDNA5 wave32, 16x16x32 bf16 -> f32 accumulate)
// ---------------------------------------------------------------------------
__device__ __forceinline__ v8f wmma_bf16(v16bf a, v16bf b, v8f c) {
  return __builtin_amdgcn_wmma_f32_16x16x32_bf16(false, a, false, b, (short)0, c,
                                                 false, false);
}

// ISA 7.12.2 fragment layout: lane L (row/col = L&15, sel = L>>4) holds
//   elems 0..7  = K = k0 + sel*8 + {0..7}
//   elems 8..15 = K = k0 + 16 + sel*8 + {0..7}
__device__ __forceinline__ v16bf frag_f32(const float* __restrict__ row, int k0, int sel) {
  const float4* p0 = reinterpret_cast<const float4*>(row + k0 + sel * 8);
  const float4* p1 = reinterpret_cast<const float4*>(row + k0 + 16 + sel * 8);
  float4 a0 = p0[0], a1 = p0[1], c0 = p1[0], c1 = p1[1];
  v16bf r;
  r[0]  = (__bf16)a0.x; r[1]  = (__bf16)a0.y; r[2]  = (__bf16)a0.z; r[3]  = (__bf16)a0.w;
  r[4]  = (__bf16)a1.x; r[5]  = (__bf16)a1.y; r[6]  = (__bf16)a1.z; r[7]  = (__bf16)a1.w;
  r[8]  = (__bf16)c0.x; r[9]  = (__bf16)c0.y; r[10] = (__bf16)c0.z; r[11] = (__bf16)c0.w;
  r[12] = (__bf16)c1.x; r[13] = (__bf16)c1.y; r[14] = (__bf16)c1.z; r[15] = (__bf16)c1.w;
  return r;
}

__device__ __forceinline__ v16bf frag_bf16(const __bf16* row, int k0, int sel) {
  v8bf lo = *reinterpret_cast<const v8bf*>(row + k0 + sel * 8);
  v8bf hi = *reinterpret_cast<const v8bf*>(row + k0 + 16 + sel * 8);
  return __builtin_shufflevector(lo, hi, 0, 1, 2, 3, 4, 5, 6, 7,
                                 8, 9, 10, 11, 12, 13, 14, 15);
}

// ---------------------------------------------------------------------------
// K0: one-shot f32 -> bf16 weight conversion (so GEMM loops are cvt-free)
// ---------------------------------------------------------------------------
__global__ __launch_bounds__(256) void k_cvt_bf16(const float* __restrict__ src,
                                                  __bf16* __restrict__ dst, int n4) {
  const int i = blockIdx.x * 256 + threadIdx.x;
  if (i >= n4) return;
  const float4 v = reinterpret_cast<const float4*>(src)[i];
  v4bf o;
  o[0] = (__bf16)v.x; o[1] = (__bf16)v.y; o[2] = (__bf16)v.z; o[3] = (__bf16)v.w;
  reinterpret_cast<v4bf*>(dst)[i] = o;
}

// ---------------------------------------------------------------------------
// K1: encoder GEMM relu(entities @ W1^T + b1) fused with masked mean pool.
// Block tile 64(M = entities of batch b) x 128(H). 8 waves, each 16x64.
// Grid (hslab, b): hslab fastest -> entities read from HBM ~once, reuse via
// L2. W1 tile (128 rows x K=64 bf16, 16 KiB) TDM-staged, double buffered:
// 8 WMMAs per wave per barrier interval.
// ---------------------------------------------------------------------------
__global__ __launch_bounds__(256) void k_encode_pool(
    const float* __restrict__ ent, const unsigned char* __restrict__ mask,
    const __bf16* __restrict__ W1b, const float* __restrict__ b1,
    float* __restrict__ pooled, __bf16* __restrict__ pooled_bf) {
  const int h0   = blockIdx.x * 128;
  const int b    = blockIdx.y;
  const int tid  = threadIdx.x;
  const int w    = tid >> 5, lane = tid & 31;
  const int mt   = w & 3, ng = w >> 2;
  const int lm   = lane & 15, sel = lane >> 4;

  __shared__ float pool[128];
  __shared__ float maskf[N_];
  __shared__ int   nvalid_s;
  if (tid < 128) pool[tid] = 0.0f;
  if (tid >= 128 && tid < 128 + N_)
    maskf[tid - 128] = mask[b * N_ + (tid - 128)] ? 1.0f : 0.0f;
  if (tid == 224) {
    int c = 0;
    for (int i = 0; i < N_; ++i) c += mask[b * N_ + i] ? 1 : 0;
    nvalid_s = c < 1 ? 1 : c;
  }

  const float* Arow = ent + ((size_t)b * N_ + mt * 16 + lm) * F_;
  v8f acc[4] = {{}, {}, {}, {}};

#if USE_TDM
  __shared__ __bf16 Btile[2][128][64];                         // 2 x 16 KiB
  const int w0s = __builtin_amdgcn_readfirstlane(tid) >> 5;    // scalar wave id
  const __bf16* Bbase = W1b + (size_t)h0 * F_;
  if (w0s == 0) {
    tdm_load_2d_bf16((unsigned)(unsigned long long)&Btile[0][0][0], Bbase,
                     F_, H_, F_, 64, 128);
    __builtin_amdgcn_s_wait_tensorcnt(0);
  }
  __syncthreads();
  for (int ks = 0; ks < F_ / 64; ++ks) {
    const int cur = ks & 1;
    const int kn  = (ks + 1) * 64;
    if (w0s == 0 && kn < F_)
      tdm_load_2d_bf16((unsigned)(unsigned long long)&Btile[cur ^ 1][0][0],
                       Bbase + kn, F_, H_, F_, 64, 128);
    __builtin_prefetch(Arow + ks * 64 + 128, 0, 1);
#pragma unroll
    for (int kk = 0; kk < 2; ++kk) {
      v16bf a = frag_f32(Arow, ks * 64 + kk * 32, sel);
      v16bf bfr[4];
#pragma unroll
      for (int nt = 0; nt < 4; ++nt)   // batch all loads under one wait
        bfr[nt] = frag_bf16(&Btile[cur][ng * 64 + nt * 16 + lm][0], kk * 32, sel);
#pragma unroll
      for (int nt = 0; nt < 4; ++nt)
        acc[nt] = wmma_bf16(a, bfr[nt], acc[nt]);
    }
    if (w0s == 0 && kn < F_) __builtin_amdgcn_s_wait_tensorcnt(0);
    __syncthreads();
  }
#else
  __syncthreads();
  for (int k0 = 0; k0 < F_; k0 += 32) {
    v16bf a = frag_f32(Arow, k0, sel);
#pragma unroll
    for (int nt = 0; nt < 4; ++nt) {
      const __bf16* Brow = W1b + (size_t)(h0 + ng * 64 + nt * 16 + lm) * F_;
      acc[nt] = wmma_bf16(a, frag_bf16(Brow, k0, sel), acc[nt]);
    }
  }
#endif

#pragma unroll
  for (int nt = 0; nt < 4; ++nt) {
    const int   col  = h0 + ng * 64 + nt * 16 + lm;
    const float bias = b1[col];
    float part = 0.0f;
#pragma unroll
    for (int j = 0; j < 8; ++j) {
      const int m = mt * 16 + j + sel * 8;       // entity row within batch b
      float v = acc[nt][j] + bias;
      v = v > 0.0f ? v : 0.0f;
      part += maskf[m] * v;                      // branch-free masking
    }
    part += __shfl_xor(part, 16, 32);            // fold the two row-halves
    if (sel == 0) atomicAdd(&pool[col - h0], part);
  }
  __syncthreads();

  if (tid < 128) {
    const float p = pool[tid] / (float)nvalid_s;
    pooled[(size_t)b * H_ + h0 + tid]    = p;
    pooled_bf[(size_t)b * H_ + h0 + tid] = (__bf16)p;
  }
}

// ---------------------------------------------------------------------------
// K2: gating — logits, softmax, top-2, renormalized weights. One block per b.
// ---------------------------------------------------------------------------
__global__ __launch_bounds__(256) void k_gate(
    const float* __restrict__ pooled, const float* __restrict__ Wg,
    const float* __restrict__ bg, float* __restrict__ gate_probs,
    int* __restrict__ topi, float* __restrict__ topw) {
  const int b    = blockIdx.x;
  const int tid  = threadIdx.x;
  const int e    = tid >> 5;
  const int lane = tid & 31;

  const float* p    = pooled + (size_t)b * H_;
  const float* wrow = Wg + (size_t)e * H_;
  float s = 0.0f;
  for (int h = lane; h < H_; h += 32) s += p[h] * wrow[h];
#pragma unroll
  for (int off = 16; off > 0; off >>= 1) s += __shfl_down(s, off, 32);

  __shared__ float lg[E_];
  if (lane == 0) lg[e] = s + bg[e];
  __syncthreads();

  if (tid == 0) {
    float mx = lg[0];
#pragma unroll
    for (int i = 1; i < E_; ++i) mx = lg[i] > mx ? lg[i] : mx;
    float ex[E_];
    float sum = 0.0f;
#pragma unroll
    for (int i = 0; i < E_; ++i) { ex[i] = __expf(lg[i] - mx); sum += ex[i]; }
    const float inv = 1.0f / sum;
    float pr[E_];
#pragma unroll
    for (int i = 0; i < E_; ++i) {
      pr[i] = ex[i] * inv;
      gate_probs[(size_t)b * E_ + i] = pr[i];
    }
    int i0 = 0; float m0 = pr[0];
#pragma unroll
    for (int i = 1; i < E_; ++i) if (pr[i] > m0) { m0 = pr[i]; i0 = i; }
    int i1 = (i0 == 0) ? 1 : 0; float m1 = pr[i1];
#pragma unroll
    for (int i = 0; i < E_; ++i) if (i != i0 && pr[i] > m1) { m1 = pr[i]; i1 = i; }
    const float t = 1.0f / (m0 + m1);
    topi[b * 2]     = i0;
    topi[b * 2 + 1] = i1;
    topw[b * 2]     = m0 * t;
    topw[b * 2 + 1] = m1 * t;
  }
}

// ---------------------------------------------------------------------------
// K3: expert layer 1: e1[e,b,:] = relu(pooled @ We1[e]^T + be1[e])
// Block tile 128(M=B) x 64(N=D); K=H. A (128xK64) and B (64xK64) bf16 tiles
// TDM-staged, double buffered (48 KiB LDS); inner loop is pure ds_load+WMMA.
// ---------------------------------------------------------------------------
__global__ __launch_bounds__(256) void k_expert1(
    const __bf16* __restrict__ pooled_bf, const __bf16* __restrict__ We1b,
    const float* __restrict__ be1, __bf16* __restrict__ e1) {
  const int b0  = blockIdx.x * 128;
  const int d0  = blockIdx.y * 64;
  const int e   = blockIdx.z;
  const int tid = threadIdx.x;
  const int w   = tid >> 5, lane = tid & 31;
  const int lm  = lane & 15, sel = lane >> 4;

  const __bf16* Abase = pooled_bf + (size_t)b0 * H_;
  const __bf16* Bbase = We1b + ((size_t)e * D_ + d0) * H_;

  v8f acc[4] = {{}, {}, {}, {}};

#if USE_TDM
  __shared__ __bf16 Atile[2][128][64];                         // 2 x 16 KiB
  __shared__ __bf16 Btile[2][64][64];                          // 2 x  8 KiB
  const int w0s = __builtin_amdgcn_readfirstlane(tid) >> 5;
  if (w0s == 0) {
    tdm_load_2d_bf16((unsigned)(unsigned long long)&Atile[0][0][0], Abase,
                     H_, B_, H_, 64, 128);
    tdm_load_2d_bf16((unsigned)(unsigned long long)&Btile[0][0][0], Bbase,
                     H_, D_, H_, 64, 64);
    __builtin_amdgcn_s_wait_tensorcnt(0);
  }
  __syncthreads();
  for (int ks = 0; ks < H_ / 64; ++ks) {
    const int cur = ks & 1;
    const int kn  = (ks + 1) * 64;
    if (w0s == 0 && kn < H_) {
      tdm_load_2d_bf16((unsigned)(unsigned long long)&Atile[cur ^ 1][0][0],
                       Abase + kn, H_, B_, H_, 64, 128);
      tdm_load_2d_bf16((unsigned)(unsigned long long)&Btile[cur ^ 1][0][0],
                       Bbase + kn, H_, D_, H_, 64, 64);
    }
#pragma unroll
    for (int kk = 0; kk < 2; ++kk) {
      v16bf a = frag_bf16(&Atile[cur][w * 16 + lm][0], kk * 32, sel);
      v16bf bfr[4];
#pragma unroll
      for (int nt = 0; nt < 4; ++nt)
        bfr[nt] = frag_bf16(&Btile[cur][nt * 16 + lm][0], kk * 32, sel);
#pragma unroll
      for (int nt = 0; nt < 4; ++nt)
        acc[nt] = wmma_bf16(a, bfr[nt], acc[nt]);
    }
    if (w0s == 0 && kn < H_) __builtin_amdgcn_s_wait_tensorcnt(0);
    __syncthreads();
  }
#else
  const __bf16* Arow = Abase + (size_t)(w * 16 + lm) * H_;
  for (int k0 = 0; k0 < H_; k0 += 32) {
    v16bf a = frag_bf16(Arow, k0, sel);
#pragma unroll
    for (int nt = 0; nt < 4; ++nt) {
      const __bf16* Brow = Bbase + (size_t)(nt * 16 + lm) * H_;
      acc[nt] = wmma_bf16(a, frag_bf16(Brow, k0, sel), acc[nt]);
    }
  }
#endif

#pragma unroll
  for (int nt = 0; nt < 4; ++nt) {
    const int   n    = d0 + nt * 16 + lm;
    const float bias = be1[e * D_ + n];
#pragma unroll
    for (int j = 0; j < 8; ++j) {
      const int m = b0 + w * 16 + j + sel * 8;
      float v = acc[nt][j] + bias;
      v = v > 0.0f ? v : 0.0f;
      e1[((size_t)e * B_ + m) * D_ + n] = (__bf16)v;   // [E,B,D] layout
    }
  }
}

// ---------------------------------------------------------------------------
// K4: expert layer 2: eout = e1[e,b,:] @ We2[e]^T + be2[e]; only the two
// gate-selected experts per row are written, into compact [B,2,A] slots
// (each slot written exactly once -> deterministic).
// ---------------------------------------------------------------------------
__global__ __launch_bounds__(256) void k_expert2(
    const __bf16* __restrict__ e1, const __bf16* __restrict__ We2b,
    const float* __restrict__ be2, const int* __restrict__ topi,
    float* __restrict__ eout2) {
  const int b0  = blockIdx.x * 128;
  const int a0  = blockIdx.y * 64;
  const int e   = blockIdx.z;
  const int tid = threadIdx.x;
  const int w   = tid >> 5, lane = tid & 31;
  const int lm  = lane & 15, sel = lane >> 4;

  const __bf16* Abase = e1 + ((size_t)e * B_ + b0) * D_;       // [E,B,D]
  const __bf16* Bbase = We2b + ((size_t)e * A_ + a0) * D_;

  v8f acc[4] = {{}, {}, {}, {}};

#if USE_TDM
  __shared__ __bf16 Atile[2][128][64];
  __shared__ __bf16 Btile[2][64][64];
  const int w0s = __builtin_amdgcn_readfirstlane(tid) >> 5;
  if (w0s == 0) {
    tdm_load_2d_bf16((unsigned)(unsigned long long)&Atile[0][0][0], Abase,
                     D_, B_, D_, 64, 128);
    tdm_load_2d_bf16((unsigned)(unsigned long long)&Btile[0][0][0], Bbase,
                     D_, A_, D_, 64, 64);
    __builtin_amdgcn_s_wait_tensorcnt(0);
  }
  __syncthreads();
  for (int ks = 0; ks < D_ / 64; ++ks) {
    const int cur = ks & 1;
    const int kn  = (ks + 1) * 64;
    if (w0s == 0 && kn < D_) {
      tdm_load_2d_bf16((unsigned)(unsigned long long)&Atile[cur ^ 1][0][0],
                       Abase + kn, D_, B_, D_, 64, 128);
      tdm_load_2d_bf16((unsigned)(unsigned long long)&Btile[cur ^ 1][0][0],
                       Bbase + kn, D_, A_, D_, 64, 64);
    }
#pragma unroll
    for (int kk = 0; kk < 2; ++kk) {
      v16bf a = frag_bf16(&Atile[cur][w * 16 + lm][0], kk * 32, sel);
      v16bf bfr[4];
#pragma unroll
      for (int nt = 0; nt < 4; ++nt)
        bfr[nt] = frag_bf16(&Btile[cur][nt * 16 + lm][0], kk * 32, sel);
#pragma unroll
      for (int nt = 0; nt < 4; ++nt)
        acc[nt] = wmma_bf16(a, bfr[nt], acc[nt]);
    }
    if (w0s == 0 && kn < D_) __builtin_amdgcn_s_wait_tensorcnt(0);
    __syncthreads();
  }
#else
  const __bf16* Arow = Abase + (size_t)(w * 16 + lm) * D_;
  for (int k0 = 0; k0 < D_; k0 += 32) {
    v16bf a = frag_bf16(Arow, k0, sel);
#pragma unroll
    for (int nt = 0; nt < 4; ++nt) {
      const __bf16* Brow = Bbase + (size_t)(nt * 16 + lm) * D_;
      acc[nt] = wmma_bf16(a, frag_bf16(Brow, k0, sel), acc[nt]);
    }
  }
#endif

  float bias[4];
#pragma unroll
  for (int nt = 0; nt < 4; ++nt) bias[nt] = be2[e * A_ + a0 + nt * 16 + lm];
#pragma unroll
  for (int j = 0; j < 8; ++j) {
    const int m  = b0 + w * 16 + j + sel * 8;
    const int t0 = topi[m * 2], t1 = topi[m * 2 + 1];
#pragma unroll
    for (int nt = 0; nt < 4; ++nt) {
      const int   n = a0 + nt * 16 + lm;
      const float v = acc[nt][j] + bias[nt];
      if (t0 == e) eout2[((size_t)m * 2 + 0) * A_ + n] = v;
      if (t1 == e) eout2[((size_t)m * 2 + 1) * A_ + n] = v;
    }
  }
}

// ---------------------------------------------------------------------------
// K5: weighted combine of the two selected-expert slots.
// ---------------------------------------------------------------------------
__global__ __launch_bounds__(256) void k_combine(
    const float* __restrict__ eout2, const float* __restrict__ topw,
    float* __restrict__ out) {
  const int idx = blockIdx.x * 256 + threadIdx.x;  // B*A
  const int b = idx >> 8;
  const int a = idx & (A_ - 1);
  out[idx] = topw[b * 2]     * eout2[((size_t)b * 2 + 0) * A_ + a] +
             topw[b * 2 + 1] * eout2[((size_t)b * 2 + 1) * A_ + a];
}

// ---------------------------------------------------------------------------
extern "C" void kernel_launch(void* const* d_in, const int* in_sizes, int n_in,
                              void* d_out, int out_size, void* d_ws, size_t ws_size,
                              hipStream_t stream) {
  const float*         ent  = (const float*)d_in[0];
  const unsigned char* mask = (const unsigned char*)d_in[1];  // jnp bool = 1 byte
  const float*         W1   = (const float*)d_in[2];
  const float*         b1   = (const float*)d_in[3];
  const float*         Wg   = (const float*)d_in[4];
  const float*         bg   = (const float*)d_in[5];
  const float*         We1  = (const float*)d_in[6];
  const float*         be1  = (const float*)d_in[7];
  const float*         We2  = (const float*)d_in[8];
  const float*         be2  = (const float*)d_in[9];
  float* out = (float*)d_out;  // [B*A action_logits | B*E gate_probs]

  char* ws = (char*)d_ws;
  float*  pooled    = (float*)(ws);                           //  4 MiB [B,H] f32
  __bf16* pooled_bf = (__bf16*)(ws + ((size_t)4  << 20));     //  2 MiB [B,H]
  __bf16* e1        = (__bf16*)(ws + ((size_t)6  << 20));     // 32 MiB [E,B,D]
  float*  eout2     = (float*)(ws + ((size_t)38 << 20));      //  2 MiB [B,2,A]
  int*    topi      = (int*)(ws + ((size_t)40 << 20));        //  8 KiB
  float*  topw      = (float*)(ws + ((size_t)40 << 20) + 8192);
  __bf16* W1b       = (__bf16*)(ws + ((size_t)41 << 20));     //  1 MiB
  __bf16* We1b      = (__bf16*)(ws + ((size_t)42 << 20));     // 32 MiB
  __bf16* We2b      = (__bf16*)(ws + ((size_t)74 << 20));     //  8 MiB

  // one-shot weight conversion to bf16
  k_cvt_bf16<<<dim3((H_ * F_ / 4) / 256), 256, 0, stream>>>(W1, W1b, H_ * F_ / 4);
  k_cvt_bf16<<<dim3((E_ * D_ * H_ / 4) / 256), 256, 0, stream>>>(We1, We1b,
                                                                 E_ * D_ * H_ / 4);
  k_cvt_bf16<<<dim3((E_ * A_ * D_ / 4) / 256), 256, 0, stream>>>(We2, We2b,
                                                                 E_ * A_ * D_ / 4);

  k_encode_pool<<<dim3(H_ / 128, B_), 256, 0, stream>>>(ent, mask, W1b, b1,
                                                        pooled, pooled_bf);
  k_gate<<<dim3(B_), 256, 0, stream>>>(pooled, Wg, bg, out + (size_t)B_ * A_,
                                       topi, topw);
  k_expert1<<<dim3(B_ / 128, D_ / 64, E_), 256, 0, stream>>>(pooled_bf, We1b,
                                                             be1, e1);
  k_expert2<<<dim3(B_ / 128, A_ / 64, E_), 256, 0, stream>>>(e1, We2b, be2,
                                                             topi, eout2);
  k_combine<<<dim3((B_ * A_) / 256), 256, 0, stream>>>(eout2, topw, out);
}
